// ImplicitInconsistencyModel_21775484191381
// MI455X (gfx1250) — matrix operand order repaired and checked by reading, not compiled, not run on hardware
//
#include <hip/hip_runtime.h>
#include <hip/hip_bf16.h>

// CDNA5 / gfx1250: wave32, WMMA 16x16x32 bf16 with fp32 accumulate.
typedef __attribute__((ext_vector_type(16))) __bf16 v16bf;
typedef __attribute__((ext_vector_type(8)))  float  v8f;

#define NEGC (-9.0e15f)   // matches reference NEG

__device__ __forceinline__ float lrelu(float x){ return x > 0.f ? x : 0.01f * x; }

// A-operand pack: lane (m + 16*half) holds K = half*8+{0..7} (elems 0..7) and
// K = 16+half*8+{0..7} (elems 8..15).  p points at arow + k0 + half*8.
__device__ __forceinline__ v16bf cvt_runA(const float* __restrict__ p){
  v16bf v;
#pragma unroll
  for (int i = 0; i < 8; ++i) v[i] = (__bf16)p[i];
#pragma unroll
  for (int i = 0; i < 8; ++i) v[8 + i] = (__bf16)p[16 + i];
  return v;
}
// B-operand pack: lane holds 16 contiguous K values of one column (row of W).
__device__ __forceinline__ v16bf cvt_run16(const float* __restrict__ p){
  v16bf v;
#pragma unroll
  for (int i = 0; i < 16; ++i) v[i] = (__bf16)p[i];
  return v;
}

// ---------------------------------------------------------------------------
// Generic GEMM: C[M x Nc] = act( A[M x K] @ W[Nc x K]^T + bias ), fp32 in/out,
// bf16 WMMA compute.  One wave per 16x16 output tile.
// CONCAT=true: row m -> req[b, n, :] (n<1024) else code[b, n-1024, :].
// ---------------------------------------------------------------------------
template<bool CONCAT>
__global__ void gemm_bf16_wmma(const float* __restrict__ A0, const float* __restrict__ A1,
                               const float* __restrict__ W,  const float* __restrict__ bias,
                               float* __restrict__ C, int K, int Nc, int act)
{
  const int lane = threadIdx.x & 31;
  const int half = lane >> 4, l15 = lane & 15;
  const long rowA = (long)blockIdx.y * 16 + l15;
  const int  colW = blockIdx.x * 16 + l15;

  const float* arow;
  if (CONCAT){
    const int b = (int)(rowA >> 11);      // / 2048
    const int n = (int)(rowA & 2047);
    arow = (n < 1024) ? (A0 + ((long)b * 1024 + n) * K)
                      : (A1 + ((long)b * 1024 + (n - 1024)) * K);
  } else {
    arow = A0 + rowA * (long)K;
  }
  const float* wrow = W + (long)colW * K;

  v8f acc = {};
  for (int k0 = 0; k0 < K; k0 += 32){
    v16bf a = cvt_runA (arow + k0 + half * 8);
    v16bf b = cvt_runA (wrow + k0 + half * 8);
    acc = __builtin_amdgcn_wmma_f32_16x16x32_bf16(false, a, false, b, (short)0, acc, false, false);
  }
  const float bc = bias[colW];
#pragma unroll
  for (int v = 0; v < 8; ++v){
    const long r = (long)blockIdx.y * 16 + v + 8 * half;
    float val = acc[v] + bc;
    if (act) val = fmaxf(val, 0.f);
    C[r * Nc + colW] = val;
  }
}

// ---------------------------------------------------------------------------
// s_src[i] = H[i,:] . a[:D];  s_dst[i] = H[i,:] . a[D:].  One wave per row.
// ---------------------------------------------------------------------------
__global__ void attn_scores(const float* __restrict__ H, const float* __restrict__ a,
                            int D, float* __restrict__ ss, float* __restrict__ sd)
{
  const int wave = threadIdx.x >> 5, lane = threadIdx.x & 31;
  const long row = (long)blockIdx.x * 8 + wave;
  const float* h = H + row * D;
  float s0 = 0.f, s1 = 0.f;
  for (int d = lane; d < D; d += 32){ float hv = h[d]; s0 += hv * a[d]; s1 += hv * a[D + d]; }
#pragma unroll
  for (int off = 16; off; off >>= 1){ s0 += __shfl_xor(s0, off, 32); s1 += __shfl_xor(s1, off, 32); }
  if (lane == 0){ ss[row] = s0; sd[row] = s1; }
}

// ---------------------------------------------------------------------------
// Pass 1: per-row online softmax stats over j = 0..2047.
// e_ij = adj>0 ? lrelu(ss_i + sd_j) : NEG.  All-masked rows -> max=NEG, sum=2048
// (uniform softmax, matching jax).  One wave per row.
// ---------------------------------------------------------------------------
__global__ void attn_stats(const float* __restrict__ adj, const float* __restrict__ ss,
                           const float* __restrict__ sd, float* __restrict__ rowmax,
                           float* __restrict__ rowinv)
{
  const int wave = threadIdx.x >> 5, lane = threadIdx.x & 31;
  const long row = (long)blockIdx.x * 8 + wave;
  const int  b = (int)(row >> 11), i = (int)(row & 2047);
  const float* arow = adj + ((long)b * 2048 + i) * 2048;
  const float* sdb  = sd  + (long)b * 2048;
  const float  si   = ss[row];

  float m = -3.4e38f, s = 0.f;
  for (int j = lane; j < 2048; j += 32){
    const float e = (arow[j] > 0.f) ? lrelu(si + sdb[j]) : NEGC;
    if (e > m){ s = s * __expf(m - e) + 1.f; m = e; } else { s += __expf(e - m); }
  }
#pragma unroll
  for (int off = 16; off; off >>= 1){
    const float m2 = __shfl_xor(m, off, 32), s2 = __shfl_xor(s, off, 32);
    const float mm = fmaxf(m, m2);
    s = s * __expf(m - mm) + s2 * __expf(m2 - mm);
    m = mm;
  }
  if (lane == 0){ rowmax[row] = m; rowinv[row] = 1.f / s; }
}

// ---------------------------------------------------------------------------
// Pass 2: out[i,:] = softmax_row(i) @ H  (flash-style).  One block per 16-row
// tile; P tile (16x32) built once in LDS in WMMA A-operand layout; each wave
// owns 32 output columns (two 16x16x32 WMMAs per j-chunk).
// ---------------------------------------------------------------------------
template<int D>
__global__ void attn_aggregate(const float* __restrict__ adj, const float* __restrict__ H,
                               const float* __restrict__ ss,  const float* __restrict__ sd,
                               const float* __restrict__ rowmax, const float* __restrict__ rowinv,
                               const int* __restrict__ total, float* __restrict__ out, int act)
{
  __shared__ __align__(32) __bf16 lds_p[512];           // 16x32 P tile, A-operand order
  __shared__ float l_ss[16], l_mx[16], l_inv[16], l_sd[32];

  const int tid  = threadIdx.x;
  const int lane = tid & 31, wave = tid >> 5;
  const long row0 = (long)blockIdx.x * 16;              // global row of tile
  const int  b    = (int)(row0 >> 11);
  const int  i0   = (int)(row0 & 2047);
  const float* adjB = adj + ((long)b * 2048 + i0) * 2048;
  const float* Hb   = H   + (long)b * 2048 * D;
  const float* sdb  = sd  + (long)b * 2048;

  if (tid < 16){ l_ss[tid] = ss[row0 + tid]; l_mx[tid] = rowmax[row0 + tid]; l_inv[tid] = rowinv[row0 + tid]; }
  __syncthreads();

  const int n0   = wave * 32;
  const int half = lane >> 4, l15 = lane & 15;
  v8f acc0 = {}, acc1 = {};

  for (int j0 = 0; j0 < 2048; j0 += 32){
    if (tid < 32) l_sd[tid] = sdb[j0 + tid];
    __syncthreads();
    // build normalized P in A-operand layout: lane (m+16*hh), elems per K runs
    for (int q = tid; q < 512; q += blockDim.x){
      const int m = q >> 5, k = q & 31;
      const float e = (adjB[(long)m * 2048 + j0 + k] > 0.f) ? lrelu(l_ss[m] + l_sd[k]) : NEGC;
      const float p = __expf(e - l_mx[m]) * l_inv[m];
      const int hh = k >> 4, el = k & 15;
      lds_p[(m + 16 * hh) * 16 + el] = (__bf16)p;
    }
    __syncthreads();

    const v16bf a = *(const v16bf*)&lds_p[lane * 16];
    v16bf b0, b1;
    const float* hc0 = Hb + (long)(j0 + half * 16) * D + n0 + l15;
#pragma unroll
    for (int i = 0; i < 16; ++i) b0[i] = (__bf16)hc0[(long)i * D];
    const float* hc1 = hc0 + 16;
#pragma unroll
    for (int i = 0; i < 16; ++i) b1[i] = (__bf16)hc1[(long)i * D];

    acc0 = __builtin_amdgcn_wmma_f32_16x16x32_bf16(false, a, false, b0, (short)0, acc0, false, false);
    acc1 = __builtin_amdgcn_wmma_f32_16x16x32_bf16(false, a, false, b1, (short)0, acc1, false, false);
    __syncthreads();   // protect lds_p / l_sd for next chunk
  }

  const int tot = total[b];
#pragma unroll
  for (int v = 0; v < 8; ++v){
    const int  mrow = v + 8 * half;
    const float mk  = ((i0 + mrow) < tot) ? 1.f : 0.f;
    const long  r   = row0 + mrow;
    float v0 = acc0[v], v1 = acc1[v];
    if (act){ v0 = fmaxf(v0, 0.f); v1 = fmaxf(v1, 0.f); }
    out[r * D + n0 + l15]      = v0 * mk;
    out[r * D + n0 + 16 + l15] = v1 * mk;
  }
}

// ---------------------------------------------------------------------------
// GRU scan (one direction), one block per batch element, 192 threads.
// Whh [192,64] cached in LDS (48 KB).  Gates torch order r|z|n.
// ---------------------------------------------------------------------------
__global__ void gru_scan(const float* __restrict__ xp, const float* __restrict__ Whh,
                         const float* __restrict__ bhh, float* __restrict__ y, int reverse)
{
  __shared__ float sW[192 * 64];
  __shared__ float sh[64], sg[192];
  const int b = blockIdx.x, tid = threadIdx.x;
  for (int i = tid; i < 192 * 64; i += 192) sW[i] = Whh[i];
  if (tid < 64) sh[tid] = 0.f;
  const float bh = bhh[tid];
  __syncthreads();

  for (int step = 0; step < 2048; ++step){
    const int t = reverse ? (2047 - step) : step;
    float acc = bh;
    const float* w = sW + tid * 64;
#pragma unroll 8
    for (int d = 0; d < 64; ++d) acc += sh[d] * w[d];
    sg[tid] = acc;
    __syncthreads();
    if (tid < 64){
      const long rb = (long)b * 2048 + t;
      const float* x = xp + rb * 192;
      const float r = 1.f / (1.f + __expf(-(x[tid]       + sg[tid])));
      const float z = 1.f / (1.f + __expf(-(x[64  + tid] + sg[64  + tid])));
      const float n = tanhf(x[128 + tid] + r * sg[128 + tid]);
      const float hn = (1.f - z) * n + z * sh[tid];
      y[rb * 64 + tid] = hn;
      sh[tid] = hn;
    }
    __syncthreads();
  }
}

// ---------------------------------------------------------------------------
// m[b,d] = (1/2048) * sum_{n<tot} gru_cat[b,n,d]   (d<64 -> gf, else gb)
// ---------------------------------------------------------------------------
__global__ void mean_reduce(const float* __restrict__ gf, const float* __restrict__ gb,
                            const int* __restrict__ total, float* __restrict__ mout)
{
  const int b = blockIdx.x, d = threadIdx.x;   // 128 threads
  const int tot = total[b];
  const float* src = (d < 64) ? (gf + (long)b * 2048 * 64 + d)
                              : (gb + (long)b * 2048 * 64 + (d - 64));
  float acc = 0.f;
  for (int n = 0; n < tot; ++n) acc += src[(long)n * 64];
  mout[b * 128 + d] = acc * (1.f / 2048.f);
}

// ---------------------------------------------------------------------------
// head: pooled = m @ W_fus^T + b_fus ; h1 = relu(pooled @ W_c1^T + b_c1);
// out = sigmoid(h1 @ W_c2^T + b_c2).  One block, 128 threads, loop over b.
// ---------------------------------------------------------------------------
__global__ void head(const float* __restrict__ m,  const float* __restrict__ Wf,
                     const float* __restrict__ bf_, const float* __restrict__ Wc1,
                     const float* __restrict__ bc1, const float* __restrict__ Wc2,
                     const float* __restrict__ bc2, float* __restrict__ out)
{
  __shared__ float sm[128], sp[64], sh1[128];
  const int tid = threadIdx.x;
  for (int b = 0; b < 8; ++b){
    sm[tid] = m[b * 128 + tid];
    __syncthreads();
    if (tid < 64){
      float a = bf_[tid];
      for (int d = 0; d < 128; ++d) a += sm[d] * Wf[tid * 128 + d];
      sp[tid] = a;
    }
    __syncthreads();
    {
      float a = bc1[tid];
      for (int j = 0; j < 64; ++j) a += sp[j] * Wc1[tid * 64 + j];
      sh1[tid] = fmaxf(a, 0.f);
    }
    __syncthreads();
    if (tid == 0){
      float a = bc2[0];
      for (int c = 0; c < 128; ++c) a += sh1[c] * Wc2[c];
      out[b] = 1.f / (1.f + __expf(-a));
    }
    __syncthreads();
  }
}

// ---------------------------------------------------------------------------
extern "C" void kernel_launch(void* const* d_in, const int* in_sizes, int n_in,
                              void* d_out, int out_size, void* d_ws, size_t ws_size,
                              hipStream_t stream)
{
  (void)in_sizes; (void)n_in; (void)out_size; (void)ws_size;

  const float* req    = (const float*)d_in[0];
  const float* code   = (const float*)d_in[1];
  const float* adj    = (const float*)d_in[2];
  const int*   total  = (const int*)  d_in[3];
  const float* W_proj = (const float*)d_in[4];
  const float* b_proj = (const float*)d_in[5];
  const float* W_g1   = (const float*)d_in[6];
  const float* b_g1   = (const float*)d_in[7];
  const float* a_g1   = (const float*)d_in[8];
  const float* W_g2   = (const float*)d_in[9];
  const float* b_g2   = (const float*)d_in[10];
  const float* a_g2   = (const float*)d_in[11];
  const float* Wih_f  = (const float*)d_in[12];
  const float* Whh_f  = (const float*)d_in[13];
  const float* bih_f  = (const float*)d_in[14];
  const float* bhh_f  = (const float*)d_in[15];
  const float* Wih_b  = (const float*)d_in[16];
  const float* Whh_b  = (const float*)d_in[17];
  const float* bih_b  = (const float*)d_in[18];
  const float* bhh_b  = (const float*)d_in[19];
  const float* W_fus  = (const float*)d_in[20];
  const float* b_fus  = (const float*)d_in[21];
  const float* W_c1   = (const float*)d_in[22];
  const float* b_c1   = (const float*)d_in[23];
  const float* W_c2   = (const float*)d_in[24];
  const float* b_c2   = (const float*)d_in[25];

  const long M = 16384;                       // B*N = 8*2048
  float* proj = (float*)d_ws;                 // [M,256]  16.8 MB
  float* H1   = proj + M * 256;               // [M,256]
  float* g1   = H1   + M * 256;               // [M,256]
  float* H2   = g1   + M * 256;               // [M,128]  8.4 MB
  float* gat  = H2   + M * 128;               // [M,128]
  float* ss   = gat  + M * 128;               // [M]
  float* sd   = ss   + M;
  float* rmx  = sd   + M;
  float* rinv = rmx  + M;
  float* mred = rinv + M;                     // [8,128]
  // reuse (lifetimes verified): proj -> xf, H1 -> xb, g1 -> gf||gb
  float* xf = proj;                           // [M,192]
  float* xb = H1;                             // [M,192]
  float* gf = g1;                             // [M,64]
  float* gb = g1 + M * 64;                    // [M,64]
  // total workspace ~67.3 MB

  // 1) proj = relu(concat(req,code) @ W_proj^T + b_proj)
  gemm_bf16_wmma<true ><<<dim3(16, 1024), 32, 0, stream>>>(req, code, W_proj, b_proj, proj, 768, 256, 1);
  // 2) GAT1: H1 = proj @ W_g1^T + b_g1
  gemm_bf16_wmma<false><<<dim3(16, 1024), 32, 0, stream>>>(proj, nullptr, W_g1, b_g1, H1, 256, 256, 0);
  attn_scores   <<<2048, 256, 0, stream>>>(H1, a_g1, 256, ss, sd);
  attn_stats    <<<2048, 256, 0, stream>>>(adj, ss, sd, rmx, rinv);
  attn_aggregate<256><<<1024, 256, 0, stream>>>(adj, H1, ss, sd, rmx, rinv, total, g1, /*relu*/1);
  // 3) GAT2: H2 = g1 @ W_g2^T + b_g2
  gemm_bf16_wmma<false><<<dim3(8, 1024), 32, 0, stream>>>(g1, nullptr, W_g2, b_g2, H2, 256, 128, 0);
  attn_scores   <<<2048, 256, 0, stream>>>(H2, a_g2, 128, ss, sd);
  attn_stats    <<<2048, 256, 0, stream>>>(adj, ss, sd, rmx, rinv);
  attn_aggregate<128><<<1024, 128, 0, stream>>>(adj, H2, ss, sd, rmx, rinv, total, gat, /*relu*/0);
  // 4) GRU input projections
  gemm_bf16_wmma<false><<<dim3(12, 1024), 32, 0, stream>>>(gat, nullptr, Wih_f, bih_f, xf, 128, 192, 0);
  gemm_bf16_wmma<false><<<dim3(12, 1024), 32, 0, stream>>>(gat, nullptr, Wih_b, bih_b, xb, 128, 192, 0);
  // 5) GRU scans (serial over T, parallel over batch x direction)
  gru_scan<<<8, 192, 0, stream>>>(xf, Whh_f, bhh_f, gf, 0);
  gru_scan<<<8, 192, 0, stream>>>(xb, Whh_b, bhh_b, gb, 1);
  // 6) masked mean-pool (fusion linear folded algebraically) + MLP head
  mean_reduce<<<8, 128, 0, stream>>>(gf, gb, total, mred);
  head<<<1, 128, 0, stream>>>(mred, W_fus, b_fus, W_c1, b_c1, W_c2, b_c2, (float*)d_out);
}
